// ExcInference_32753420600141
// MI455X (gfx1250) — matrix operand order repaired and checked by reading, not compiled, not run on hardware
//
#include <hip/hip_runtime.h>

// ---------------------------------------------------------------------------
// ExcInference on MI455X (gfx1250, wave32).
// Pass 1: per-token 512x257x512 Hankel GEMM (~138 GFLOP total) via bf16x3
//   split on v_wmma_f32_16x16x32_bf16. 4-way M-register-blocking per wave;
//   W slabs (64x512 bf16 hi/lo = 128 KB) staged into LDS by the Tensor Data
//   Mover (tensor_load_to_lds + s_wait_tensorcnt). Energy + argmax fused.
// Pass 2: recompute winning h row, mask, top-128, 256-row decode slice.
// ---------------------------------------------------------------------------

typedef __attribute__((ext_vector_type(16))) __bf16 v16bf;
typedef __attribute__((ext_vector_type(8)))  float  v8f;
typedef __attribute__((ext_vector_type(4)))  unsigned tdm_g0_t;
typedef __attribute__((ext_vector_type(8)))  int      tdm_g1_t;
typedef __attribute__((ext_vector_type(4)))  int      tdm_g4_t;
typedef __attribute__((ext_vector_type(8)))  int      tdm_g8_t;

#define IDIM   256
#define HDIM   512
#define WIN    512   // 2*IDIM
#define NTOK   1024  // B*T
#define SLABB  65536 // 64 rows x 512 cols x 2B

union FragU { unsigned u[8]; v16bf v; };

__device__ __forceinline__ unsigned pack2(__bf16 a, __bf16 b) {
  unsigned short ua = __builtin_bit_cast(unsigned short, a);
  unsigned short ub = __builtin_bit_cast(unsigned short, b);
  return (unsigned)ua | ((unsigned)ub << 16);
}

__device__ __forceinline__ v16bf loadA(const __bf16* p) {  // two aligned 16B runs
  FragU f;
  uint4 a = *(const uint4*)(p);
  uint4 b = *(const uint4*)(p + 16);
  f.u[0] = a.x; f.u[1] = a.y; f.u[2] = a.z; f.u[3] = a.w;
  f.u[4] = b.x; f.u[5] = b.y; f.u[6] = b.z; f.u[7] = b.w;
  return f.v;
}

__device__ __forceinline__ v16bf loadB(const unsigned* s, int base) {
  FragU f;
#pragma unroll
  for (int i = 0; i < 8; ++i) f.u[i] = s[base + i + (i >= 4 ? 4 : 0)];
  return f.v;
}

__device__ __forceinline__ void mma3(v8f& acc, v16bf ah, v16bf al,
                                     v16bf bh, v16bf bl) {
  // bf16x3 split: (Ah+Al)(Bh+Bl) ~= AhBh + AhBl + AlBh
  acc = __builtin_amdgcn_wmma_f32_16x16x32_bf16(false, ah, false, bh,
                                                (short)0, acc, false, false);
  acc = __builtin_amdgcn_wmma_f32_16x16x32_bf16(false, ah, false, bl,
                                                (short)0, acc, false, false);
  acc = __builtin_amdgcn_wmma_f32_16x16x32_bf16(false, al, false, bh,
                                                (short)0, acc, false, false);
}

// ---- prep: split fp32 W_enc into bf16 hi/lo (read many times in pass 1) ----
__global__ void __launch_bounds__(256)
split_wenc_kernel(const float* __restrict__ W,
                  __bf16* __restrict__ Whi, __bf16* __restrict__ Wlo) {
  int i = blockIdx.x * 256 + threadIdx.x;            // 512*512 = 262144 exact
  float w  = W[i];
  __bf16 h = (__bf16)w;
  Whi[i] = h;
  Wlo[i] = (__bf16)(w - (float)h);
}

// ---- pass 1: per-token energy over all 257 shifts + argmax ----------------
__global__ void __launch_bounds__(256)
energy_argmax_kernel(const float* __restrict__ x,
                     const __bf16* __restrict__ Whi,
                     const __bf16* __restrict__ Wlo,
                     const float* __restrict__ b_enc,
                     int* __restrict__ hInd) {
  // Dynamic LDS: current 64-row W slab, bf16 hi + lo (TDM destination).
  extern __shared__ __align__(16) char smem[];
  __bf16* slabHi = (__bf16*)smem;
  __bf16* slabLo = (__bf16*)(smem + SLABB);

  // xp (padded window source) staged as bf16 hi/lo, pair-packed at even AND
  // odd element phase so each lane's two 8-element K-runs are dword-aligned.
  __shared__ unsigned xph0[408], xph1[408], xpl0[408], xpl1[408];
  __shared__ float bEnc[HDIM];
  __shared__ float energy[272];
  __shared__ float rv[256];
  __shared__ int   ri[256];

  const int tid  = threadIdx.x;
  const int bt   = blockIdx.x;
  const int lane = tid & 31;
  const int wave = tid >> 5;
  const float* xr = x + bt * IDIM;

  for (int d = tid; d < 408; d += 256) {
    const int j0 = 2 * d;
    float f0 = (j0     >= 256 && j0     < 512) ? xr[j0     - 256] : 0.f;
    float f1 = (j0 + 1 >= 256 && j0 + 1 < 512) ? xr[j0 + 1 - 256] : 0.f;
    float f2 = (j0 + 2 >= 256 && j0 + 2 < 512) ? xr[j0 + 2 - 256] : 0.f;
    __bf16 h0 = (__bf16)f0, h1 = (__bf16)f1, h2 = (__bf16)f2;
    __bf16 l0 = (__bf16)(f0 - (float)h0);
    __bf16 l1 = (__bf16)(f1 - (float)h1);
    __bf16 l2 = (__bf16)(f2 - (float)h2);
    xph0[d] = pack2(h0, h1);  xph1[d] = pack2(h1, h2);
    xpl0[d] = pack2(l0, l1);  xpl1[d] = pack2(l1, l2);
  }
  for (int i = tid; i < HDIM; i += 256) bEnc[i] = b_enc[i];
  for (int i = tid; i < 272;  i += 256) energy[i] = 0.f;

  const unsigned ldsHi = (unsigned)(size_t)(void*)slabHi;
  const unsigned ldsLo = (unsigned)(size_t)(void*)slabLo;

  // 8 M-groups of 64 rows; per group: TDM-stage W slab, 8 waves split the
  // 17 N-tiles, each wave register-blocks 4 M-tiles against one B fragment.
  for (int mtg = 0; mtg < 8; ++mtg) {
    __syncthreads();   // previous slab fully consumed / fills visible
    if (wave == 0) {
      const unsigned long long gaHi =
          (unsigned long long)(size_t)Whi + (unsigned long long)mtg * SLABB;
      const unsigned long long gaLo =
          (unsigned long long)(size_t)Wlo + (unsigned long long)mtg * SLABB;
      // D# group1 (ISA 8.4): data_size=2B; tensor_dim0=512, tensor_dim1=512;
      // tile_dim0=512, tile_dim1=64; tensor_dim0_stride=512 elements.
      const tdm_g1_t g1 = { (int)(1u << 16),           // data_size=2B
                            (int)(512u << 16),          // tensor_dim0 lo16
                            (int)(512u << 16),          // td0 hi16 | td1 lo16
                            (int)(512u << 16),          // td1 hi16 | tile_dim0
                            64,                         // tile_dim1 | tile_dim2=0
                            512,                        // tensor_dim0_stride lo32
                            0, 0 };
      const tdm_g4_t gz4 = { 0, 0, 0, 0 };
      const tdm_g8_t gz8 = { 0, 0, 0, 0, 0, 0, 0, 0 };
      // D# group0 (ISA 8.3): count=1, lds_addr, 57b global_addr, type=2.
      const tdm_g0_t g0h = { 1u, ldsHi, (unsigned)gaHi,
          (unsigned)((gaHi >> 32) & 0x01FFFFFFull) | (2u << 30) };
      const tdm_g0_t g0l = { 1u, ldsLo, (unsigned)gaLo,
          (unsigned)((gaLo >> 32) & 0x01FFFFFFull) | (2u << 30) };
      __builtin_amdgcn_tensor_load_to_lds(g0h, g1, gz4, gz4, gz8, 0);
      __builtin_amdgcn_tensor_load_to_lds(g0l, g1, gz4, gz4, gz8, 0);
      __builtin_amdgcn_s_wait_tensorcnt(0);
    }
    __syncthreads();   // slab visible to all waves

    for (int nt = wave; nt < 17; nt += 8) {          // wave-uniform bound
      const int scol = nt * 16 + (lane & 15);        // B: lane -> shift column
      const int koff = (lane >= 16) ? 8 : 0;         // A K-phase per half-wave
      const int boff = (lane >= 16) ? 8 : 0;         // B K-phase per half-wave
      const __bf16* arh = slabHi + (size_t)(lane & 15) * WIN + koff;
      const __bf16* arl = slabLo + (size_t)(lane & 15) * WIN + koff;

      v8f acc[4] = {{}, {}, {}, {}};
#pragma unroll 2
      for (int kk = 0; kk < WIN; kk += 32) {
        // B fragment: two 8-element xp runs 16 apart; pick even/odd-phase packed
        // copy so the 8 dword LDS reads are aligned. Shared by 4 M-tiles.
        const int t0 = scol + kk + boff;
        const unsigned* shp = (t0 & 1) ? xph1 : xph0;
        const unsigned* slp = (t0 & 1) ? xpl1 : xpl0;
        const int base = t0 >> 1;
        const v16bf bh = loadB(shp, base);
        const v16bf bl = loadB(slp, base);
#pragma unroll
        for (int m4 = 0; m4 < 4; ++m4) {
          const v16bf ah = loadA(arh + (size_t)(m4 * 16) * WIN + kk);
          const v16bf al = loadA(arl + (size_t)(m4 * 16) * WIN + kk);
          mma3(acc[m4], ah, al, bh, bl);
        }
      }
      // C layout: lane n in {0..15} holds M = j (VGPR j), lanes 16..31 M=8+j.
#pragma unroll
      for (int m4 = 0; m4 < 4; ++m4) {
        const int mbase = mtg * 64 + m4 * 16 + ((lane >= 16) ? 8 : 0);
        float e = 0.f;
#pragma unroll
        for (int j = 0; j < 8; ++j) {
          float v = acc[m4][j] + bEnc[mbase + j];
          e += v * v;
        }
        atomicAdd(&energy[nt * 16 + (lane & 15)], e);   // ds_add_f32
      }
    }
  }
  __syncthreads();

  // argmax over shifts 0..256 (first-max tie-break like jnp.argmax)
  float v  = energy[tid];
  int   ix = tid;
  if (tid == 0) {
    float v2 = energy[256];
    if (v2 > v) { v = v2; ix = 256; }
  }
  rv[tid] = v; ri[tid] = ix;
  __syncthreads();
  for (int s = 128; s > 0; s >>= 1) {
    if (tid < s) {
      float vo = rv[tid + s]; int io = ri[tid + s];
      if (vo > rv[tid] || (vo == rv[tid] && io < ri[tid])) {
        rv[tid] = vo; ri[tid] = io;
      }
    }
    __syncthreads();
  }
  if (tid == 0) hInd[bt] = ri[0];
}

// ---- pass 2: recompute winning h row, mask, top-128, decode slice ---------
__global__ void __launch_bounds__(256)
decode_kernel(const float* __restrict__ x, const int* __restrict__ mask_prev,
              const float* __restrict__ W_enc, const float* __restrict__ b_enc,
              const float* __restrict__ W_dec, const float* __restrict__ b_dec,
              const int* __restrict__ hInd, float* __restrict__ out) {
  __shared__ float xrow[IDIM];
  __shared__ float hv[HDIM];
  __shared__ float hsel[HDIM];
  const int tid = threadIdx.x;
  const int bt  = blockIdx.x;
  const int s   = hInd[bt];

  xrow[tid] = x[bt * IDIM + tid];
  __syncthreads();

  const int w0 = IDIM - s;       // first window index hitting real (unpadded) x
  for (int m = tid; m < HDIM; m += 256) {
    float acc = b_enc[m];
    const float* wr = W_enc + (size_t)m * WIN;
#pragma unroll 4
    for (int w = 0; w < IDIM; ++w) acc += xrow[w] * wr[w0 + w];
    if (mask_prev[bt * HDIM + m] > 0) acc = 0.f;
    hv[m] = acc;
  }
  __syncthreads();

  // keep top-128 by h^2 (ties -> lower index, matching lax.top_k)
  for (int m = tid; m < HDIM; m += 256) {
    const float v = hv[m] * hv[m];
    int rank = 0;
    for (int mm = 0; mm < HDIM; ++mm) {
      const float vv = hv[mm] * hv[mm];
      rank += (vv > v) || (vv == v && mm < m);
    }
    hsel[m] = (rank < 128) ? hv[m] : 0.f;
  }
  __syncthreads();

  // only the 256 needed decoder rows: o = 256 - s + i
  const int o = IDIM - s + tid;
  float acc = b_dec[o];
  const float* wd = W_dec + (size_t)o * HDIM;
#pragma unroll 4
  for (int hh = 0; hh < HDIM; ++hh) acc += hsel[hh] * wd[hh];
  out[bt * IDIM + tid] = acc;
}

// ---------------------------------------------------------------------------
extern "C" void kernel_launch(void* const* d_in, const int* in_sizes, int n_in,
                              void* d_out, int out_size, void* d_ws, size_t ws_size,
                              hipStream_t stream) {
  const float* x         = (const float*)d_in[0];
  const int*   mask_prev = (const int*)  d_in[1];
  const float* W_enc     = (const float*)d_in[2];
  const float* b_enc     = (const float*)d_in[3];
  const float* W_dec     = (const float*)d_in[4];
  const float* b_dec     = (const float*)d_in[5];
  float*       out       = (float*)d_out;

  char* ws = (char*)d_ws;
  __bf16* Whi = (__bf16*)(ws);                       // 512 KB
  __bf16* Wlo = (__bf16*)(ws + (size_t)524288);      // 512 KB
  int*    hIx = (int*)   (ws + (size_t)1048576);     // 4 KB

  split_wenc_kernel<<<1024, 256, 0, stream>>>(W_enc, Whi, Wlo);
  // 128 KB dynamic LDS for the TDM-staged W slab (WGP has 320 KB).
  energy_argmax_kernel<<<NTOK, 256, 2 * SLABB, stream>>>(x, Whi, Wlo, b_enc, hIx);
  decode_kernel<<<NTOK, 256, 0, stream>>>(x, mask_prev, W_enc, b_enc,
                                          W_dec, b_dec, hIx, out);
}
